// Attention_59459527246596
// MI455X (gfx1250) — matrix operand order repaired and checked by reading, not compiled, not run on hardware
//
#include <hip/hip_runtime.h>
#include <math.h>

// Problem constants (from reference)
#define B_  16
#define TQ_ 2048
#define TC_ 2048
#define H_  1024
#define H2_ 2048
#define HC  256          // H-slice accumulated per flash block (16 wmma tiles)
#define CP  1032         // Clds pitch (ushorts)  -> 2064 B rows, 16B aligned
#define TP  40           // CldsT pitch (ushorts) -> 80 B rows
#define PP  40           // Plds pitch

typedef __attribute__((ext_vector_type(16))) __bf16          v16bf;
typedef __attribute__((ext_vector_type(16))) unsigned short  v16us;
typedef __attribute__((ext_vector_type(8)))  unsigned short  v8us;
typedef __attribute__((ext_vector_type(4)))  unsigned short  v4us;
typedef __attribute__((ext_vector_type(8)))  float           v8f;

// ---------------------------------------------------------------- helpers ---
static __device__ __forceinline__ unsigned short f2bf(float f) {
  unsigned int u = __builtin_bit_cast(unsigned int, f);
  u += 0x7FFFu + ((u >> 16) & 1u);        // round-to-nearest-even
  return (unsigned short)(u >> 16);
}

static __device__ __forceinline__ v8f wmma_bf16(v16us a, v16us b, v8f c) {
  // D = A(16x32 bf16) * B(32x16 bf16) + C(f32)
  return __builtin_amdgcn_wmma_f32_16x16x32_bf16(
      false, __builtin_bit_cast(v16bf, a),
      false, __builtin_bit_cast(v16bf, b),
      (short)0, c, false, false);
}

// A-matrix 16x32 fragment from a bf16 row (global or LDS).
// lane half h: elems 0..15 -> K = 16h..16h+15
static __device__ __forceinline__ v16us afrag_bf16(const unsigned short* rowbase, int half) {
  v8us lo = *(const v8us*)(rowbase + 16 * half);
  v8us hi = *(const v8us*)(rowbase + 16 * half + 8);
  v16us r;
#pragma unroll
  for (int i = 0; i < 8; ++i) { r[i] = lo[i]; r[8 + i] = hi[i]; }
  return r;
}

// Async DMA: 16B global -> LDS, tracked by ASYNCcnt (CDNA5 path).
static __device__ __forceinline__ void async_g2l_b128(unsigned lds_off, const unsigned short* g) {
  asm volatile("global_load_async_to_lds_b128 %0, %1, off" :: "v"(lds_off), "v"(g) : "memory");
}
static __device__ __forceinline__ void wait_async0() {
  asm volatile("s_wait_asynccnt 0" ::: "memory");
}

// Stage C[kc..kc+31][0..H) (bf16) into LDS row-major with pitch CP, via async DMA.
// 64 KB chunk = 4096 x 16B transfers / 256 threads = 16 per thread.
static __device__ __forceinline__ void stage_C_async(const unsigned short* Cb, int kc,
                                                     unsigned lds_byte_base, int tid) {
#pragma unroll
  for (int j = 0; j < 16; ++j) {
    int k  = tid + 256 * j;
    int r  = k >> 7;                  // ctx row 0..31
    int c8 = (k & 127) * 8;           // 8-ushort column group
    const unsigned short* g = Cb + (size_t)(kc + r) * H_ + c8;
    unsigned lo = lds_byte_base + (unsigned)((r * CP + c8) * 2);
    async_g2l_b128(lo, g);
  }
}

// Stage transposed C slice CldsT[h - h1][ctx] for the mix B operand
// (normal loads + ds stores; overlaps compute via double buffering).
static __device__ __forceinline__ void stage_CT(const unsigned short* Cb, int kc, int h1,
                                                unsigned short* CldsT, int tid) {
  const int kk  = tid >> 3;            // ctx row 0..31
  const int hh0 = (tid & 7) * 32;      // 32 h-columns per thread
  const unsigned short* g = Cb + (size_t)(kc + kk) * H_ + h1 + hh0;
#pragma unroll
  for (int jj = 0; jj < 4; ++jj) {
    v8us x = *(const v8us*)(g + jj * 8);
#pragma unroll
    for (int e = 0; e < 8; ++e) CldsT[(hh0 + jj * 8 + e) * TP + kk] = x[e];
  }
}

// ------------------------------------------------------------ kernel 0 ------
// One-shot f32 -> bf16 conversion (bandwidth-bound).
__global__ void __launch_bounds__(256)
cvt_bf16_kernel(const float* __restrict__ src, unsigned short* __restrict__ dst, int n4) {
  int i = blockIdx.x * 256 + threadIdx.x;
  if (i >= n4) return;
  float4 x = ((const float4*)src)[i];
  v4us o;
  o[0] = f2bf(x.x); o[1] = f2bf(x.y); o[2] = f2bf(x.z); o[3] = f2bf(x.w);
  *(v4us*)(dst + (size_t)i * 4) = o;
}

// ------------------------------------------------------------ kernel 1 ------
// Fused flash attention: online softmax + mix accumulation for a 256-wide H
// slice. Block = 8 waves, each owning a distinct 16-row q tile; C chunks are
// double-buffered in LDS and fetched with the async tensor/LDS DMA path.
__global__ void __launch_bounds__(256)
attn_flash_kernel(const unsigned short* __restrict__ Qb,
                  const unsigned short* __restrict__ Call,
                  unsigned short* __restrict__ Mixb) {
  extern __shared__ unsigned short smem[];
  // layout: Clds[2][32][CP] | CldsT[2][HC][TP] | Plds[8][16][PP]
  unsigned short* CldsTb = smem + 2 * 32 * CP;
  unsigned short* Plds   = CldsTb + 2 * HC * TP;

  const int tid  = threadIdx.x;
  const int lane = tid & 31, wave = tid >> 5;
  const int half = lane >> 4, ml = lane & 15;
  const int rowtile = blockIdx.x * 8 + wave;
  const int b  = rowtile / (TQ_ / 16);
  const int q0 = (rowtile % (TQ_ / 16)) * 16;
  const int h1 = blockIdx.y * HC;
  const unsigned short* Qrow = Qb + ((size_t)b * TQ_ + q0 + ml) * H_;
  const unsigned short* Cb   = Call + (size_t)b * TC_ * H_;
  unsigned short* Pw = Plds + wave * 16 * PP;
  const unsigned smembase = (unsigned)(size_t)smem;

  float rm[8], rs[8];
#pragma unroll
  for (int v = 0; v < 8; ++v) { rm[v] = -3.0e38f; rs[v] = 0.f; }
  v8f acc[HC / 16];
#pragma unroll
  for (int t = 0; t < HC / 16; ++t) acc[t] = (v8f){};

  // prologue: prefetch chunk 0 into buffer 0
  stage_C_async(Cb, 0, smembase, tid);
  stage_CT(Cb, 0, h1, CldsTb, tid);

  for (int kc = 0; kc < TC_; kc += 32) {
    const int cur = (kc >> 5) & 1;
    wait_async0();
    __syncthreads();                       // chunk `cur` fully staged, all waves done with `nxt`
    if (kc + 32 < TC_) {
      const int nxt = cur ^ 1;
      stage_C_async(Cb, kc + 32, smembase + (unsigned)(nxt * 32 * CP * 2), tid);
      stage_CT(Cb, kc + 32, h1, CldsTb + nxt * HC * TP, tid);
    }
    const unsigned short* Cc = smem + cur * 32 * CP;
    const unsigned short* Ct = CldsTb + cur * HC * TP;

    // scores: 16 q-rows x 32 ctx-cols, contraction over H
    v8f s0 = {}, s1 = {};
    for (int h0 = 0; h0 < H_; h0 += 32) {
      v16us a  = afrag_bf16(Qrow + h0, half);
      v16us b0 = *(const v16us*)(Cc + ml        * CP + h0 + 16 * half);
      v16us b1 = *(const v16us*)(Cc + (16 + ml) * CP + h0 + 16 * half);
      s0 = wmma_bf16(a, b0, s0);
      s1 = wmma_bf16(a, b1, s1);
    }

    // online softmax: new max, exp, running sum, accumulator rescale factor
    float f[8];
#pragma unroll
    for (int v = 0; v < 8; ++v) {
      float tmax = fmaxf(s0[v], s1[v]);
#pragma unroll
      for (int m = 1; m < 16; m <<= 1) tmax = fmaxf(tmax, __shfl_xor(tmax, m));
      float nm = fmaxf(rm[v], tmax);
      float p0 = __expf(s0[v] - nm), p1 = __expf(s1[v] - nm);
      float es = p0 + p1;
#pragma unroll
      for (int m = 1; m < 16; m <<= 1) es += __shfl_xor(es, m);
      f[v]  = __expf(rm[v] - nm);
      rs[v] = rs[v] * f[v] + es;
      rm[v] = nm;
      const int r = v + 8 * half;
      Pw[r * PP + ml]      = f2bf(p0);     // D-layout -> row-major P staging
      Pw[r * PP + 16 + ml] = f2bf(p1);
    }
#pragma unroll
    for (int t = 0; t < HC / 16; ++t)
#pragma unroll
      for (int v = 0; v < 8; ++v) acc[t][v] *= f[v];

    // mix accumulate: P(16x32) x C(32xHC)
    v16us a = afrag_bf16(Pw + ml * PP, half);
#pragma unroll
    for (int t = 0; t < HC / 16; ++t) {
      v16us bt = *(const v16us*)(Ct + (t * 16 + ml) * TP + 16 * half);
      acc[t] = wmma_bf16(a, bt, acc[t]);
    }
  }

  // epilogue: normalize by 1/l, store bf16 mix
  float li[8];
#pragma unroll
  for (int v = 0; v < 8; ++v) li[v] = 1.0f / rs[v];
#pragma unroll
  for (int t = 0; t < HC / 16; ++t)
#pragma unroll
    for (int v = 0; v < 8; ++v)
      Mixb[((size_t)b * TQ_ + q0 + v + 8 * half) * H_ + h1 + t * 16 + ml] =
          f2bf(acc[t][v] * li[v]);
}

// ------------------------------------------------------------ kernel 2 ------
// out = tanh([mix | Q] @ W^T), all operands pre-converted bf16, no LDS needed
// (A rows are L2-resident and shared across waves; W is 4 MB bf16, L2-resident).
__global__ void __launch_bounds__(256)
attn_proj_kernel(const unsigned short* __restrict__ Mixb,
                 const unsigned short* __restrict__ Qb,
                 const unsigned short* __restrict__ Wb,
                 float* __restrict__ Out) {
  const int tid  = threadIdx.x;
  const int lane = tid & 31, wave = tid >> 5;
  const int half = lane >> 4, ml = lane & 15;
  const int b  = blockIdx.x / (TQ_ / 32);
  const int q0 = (blockIdx.x % (TQ_ / 32)) * 32;
  const int qs = wave & 1;                   // q sub-tile (16 rows)
  const int hb = (wave >> 1) * 256;          // 256-wide output slice
  const size_t rowoff = ((size_t)b * TQ_ + q0 + qs * 16 + ml) * H_;
  const unsigned short* mrow = Mixb + rowoff;
  const unsigned short* qrow = Qb + rowoff;

  v8f acc[16];
#pragma unroll
  for (int t = 0; t < 16; ++t) acc[t] = (v8f){};

  for (int d0 = 0; d0 < H_; d0 += 32) {      // first half of K: mix part
    v16us a = afrag_bf16(mrow + d0, half);
#pragma unroll
    for (int t = 0; t < 16; ++t) {
      v16us bt = *(const v16us*)(Wb + (size_t)(hb + t * 16 + ml) * H2_ + d0 + 16 * half);
      acc[t] = wmma_bf16(a, bt, acc[t]);
    }
  }
  for (int d0 = 0; d0 < H_; d0 += 32) {      // second half of K: query part
    v16us a = afrag_bf16(qrow + d0, half);
#pragma unroll
    for (int t = 0; t < 16; ++t) {
      v16us bt = *(const v16us*)(Wb + (size_t)(hb + t * 16 + ml) * H2_ + H_ + d0 + 16 * half);
      acc[t] = wmma_bf16(a, bt, acc[t]);
    }
  }

#pragma unroll
  for (int t = 0; t < 16; ++t)
#pragma unroll
    for (int v = 0; v < 8; ++v)
      Out[((size_t)b * TQ_ + q0 + qs * 16 + v + 8 * half) * H_ + hb + t * 16 + ml] =
          tanhf(acc[t][v]);
}

// ------------------------------------------------------------- launcher -----
extern "C" void kernel_launch(void* const* d_in, const int* in_sizes, int n_in,
                              void* d_out, int out_size, void* d_ws, size_t ws_size,
                              hipStream_t stream) {
  (void)in_sizes; (void)n_in; (void)out_size; (void)ws_size;
  const float* Q = (const float*)d_in[0];
  const float* C = (const float*)d_in[1];
  const float* W = (const float*)d_in[2];
  float* out = (float*)d_out;

  // workspace layout (bf16): Qb | Cb | Wb | Mixb   (~196 MB)
  unsigned short* Qb   = (unsigned short*)d_ws;
  unsigned short* Cb   = Qb + (size_t)B_ * TQ_ * H_;
  unsigned short* Wb   = Cb + (size_t)B_ * TC_ * H_;
  unsigned short* Mixb = Wb + (size_t)H_ * H2_;

  const int nQ4 = B_ * TQ_ * H_ / 4;   // 8388608
  const int nW4 = H_ * H2_ / 4;        // 524288
  cvt_bf16_kernel<<<dim3(nQ4 / 256), dim3(256), 0, stream>>>(Q, Qb, nQ4);
  cvt_bf16_kernel<<<dim3(nQ4 / 256), dim3(256), 0, stream>>>(C, Cb, nQ4);
  cvt_bf16_kernel<<<dim3(nW4 / 256), dim3(256), 0, stream>>>(W, Wb, nW4);

  const size_t shm_flash =
      (size_t)(2 * 32 * CP + 2 * HC * TP + 8 * 16 * PP) * sizeof(unsigned short); // ~179 KB
  hipFuncSetAttribute(reinterpret_cast<const void*>(attn_flash_kernel),
                      hipFuncAttributeMaxDynamicSharedMemorySize, (int)shm_flash);

  attn_flash_kernel<<<dim3(B_ * TQ_ / 128, H_ / HC), dim3(256), shm_flash, stream>>>(Qb, Cb, Mixb);
  attn_proj_kernel<<<dim3(B_ * (TQ_ / 32)), dim3(256), 0, stream>>>(Mixb, Qb, Wb, out);
}
